// MPS_RNN_1D_8297876816278
// MI455X (gfx1250) — compile-verified, matrix operand order checked by reading, not run on hardware
//
#include <hip/hip_runtime.h>
#include <hip/hip_bf16.h>
#include <math.h>

// ---------------------------------------------------------------------------
// MPS-RNN 1D scan, CDNA5 (gfx1250, wave32).
//
//   - Per site, the complex matvec  h_sel(6c) x M(6c x 24c)  is a real GEMM
//     (batch x 13) x (13 x 48) run as v_wmma_f32_16x16x32_f16
//     (M=16 batches, N=16 of 48 comps, K: 12 data + 1 bias row, pad to 32).
//   - One wave owns 32 batches (2 M-tiles) -> 6 WMMAs per site, all B tiles
//     loaded up-front in one clause so loads overlap the LDS staging.
//   - Bias folded into B row K=12 (A supplies constant 1.0h) so C is inline 0.
//   - h_sel published to LDS in f16 -> A fragments are pure LDS loads.
//   - Wave-private LDS ping-pong fenced with s_wait_dscnt (no block barriers).
//   - P normalization algebraically reduced: amp *= sqrt(acc_q / sum(acc)).
//   - Phase accumulated as a unit complex product (no atan2/sincos).
// ---------------------------------------------------------------------------

typedef __attribute__((ext_vector_type(16))) _Float16 v16h;
typedef __attribute__((ext_vector_type(8)))  float    v8f;

#define NSITES   48
#define DCUT_    6
#define HL_      4
#define NBATCH_  65536
#define WAVES    8          // waves per block (independent)
#define BATCH_PER_WAVE 32   // 2 WMMA M-tiles
#define NCOMP    48         // 4*6 complex -> 48 reals

// workspace layout (bytes)
#define WS_BPACK_OFF  0
#define WS_BPACK_SZ   (NSITES * 3 * 32 * 16 * 2)       // 147456
#define WS_PRM_OFF    (WS_BPACK_OFF + WS_BPACK_SZ)
#define WS_PRM_SZ     (NSITES * 24 * 4)                // e2[6] wre[6] wim[6] c[2] pad[4]

// wave-local LDS fence: this wave's DS stores complete before following DS
// loads; "memory" clobber stops compiler reordering (global loads placed
// before the fence still overlap, their waitcnt lands at first use).
__device__ __forceinline__ void wave_lds_fence() {
    asm volatile("s_wait_dscnt 0x0" ::: "memory");
}

// ---------------------------------------------------------------------------
// Pre-pass: pack transfer matrices (+bias row) into the WMMA f16 B-operand
// register layout, and site params into one 24-float row.
//
// K convention: K=0..5 -> Re(h_sel[a]), K=6..11 -> Im(h_sel[a]), K=12 -> 1.0
// (bias). Output component n = (i*6+b)*2 + (0:re,1:im).
//   re-out:  K=a -> Re M[i,a,b],  K=6+a -> -Im M[i,a,b],  K=12 -> Re v[i,b]
//   im-out:  K=a -> Im M[i,a,b],  K=6+a ->  Re M[i,a,b],  K=12 -> Im v[i,b]
// B register layout (16-bit operand, wave32, ISA 7.12.2):
//   lane -> N = ntile*16 + (lane & 15)
//   K    = e + (e>=8 ? 8 : 0) + (lane>=16 ? 8 : 0),  e = half index 0..15
// ---------------------------------------------------------------------------
__global__ void mpsrnn_pack_params(const float* __restrict__ M,
                                   const float* __restrict__ v,
                                   const float* __restrict__ w,
                                   const float* __restrict__ c,
                                   const float* __restrict__ eta,
                                   _Float16* __restrict__ Bpack,
                                   float* __restrict__ prm)
{
    const int s   = blockIdx.x;          // site
    const int tid = threadIdx.x;

    for (int idx = tid; idx < 3 * 32 * 16; idx += blockDim.x) {
        const int e    = idx & 15;
        const int lane = (idx >> 4) & 31;
        const int nt   = idx >> 9;
        const int K    = e + (e >= 8 ? 8 : 0) + (lane >= 16 ? 8 : 0);
        const int n    = nt * 16 + (lane & 15);
        float val = 0.0f;
        if (n < NCOMP) {
            const int  i   = (n >> 1) / 6;
            const int  b   = (n >> 1) % 6;
            const bool nim = (n & 1);
            if (K < 12) {
                const int  a   = (K >= 6) ? (K - 6) : K;
                const bool kim = (K >= 6);
                const int  mi  = (((s * HL_ + i) * DCUT_ + a) * DCUT_ + b) * 2;
                const float mre = M[mi + 0];
                const float mim = M[mi + 1];
                val = nim ? (kim ? mre : mim) : (kim ? -mim : mre);
            } else if (K == 12) {        // bias row (A supplies 1.0 here)
                val = v[((s * HL_ + i) * DCUT_ + b) * 2 + (nim ? 1 : 0)];
            }
        }
        Bpack[(s * 3 + nt) * 512 + lane * 16 + e] = (_Float16)val;
    }

    // per-site param row: [e2 x6][wre x6][wim x6][cre][cim][pad x4]
    if (tid < 24) {
        float val = 0.0f;
        if (tid < 6) {
            const float er = eta[(s * DCUT_ + tid) * 2 + 0];
            const float ei = eta[(s * DCUT_ + tid) * 2 + 1];
            val = er * er + ei * ei;
        } else if (tid < 12)  val = w[(s * DCUT_ + (tid - 6))  * 2 + 0];
        else if   (tid < 18)  val = w[(s * DCUT_ + (tid - 12)) * 2 + 1];
        else if   (tid == 18) val = c[s * 2 + 0];
        else if   (tid == 19) val = c[s * 2 + 1];
        prm[s * 24 + tid] = val;
    }
}

// ---------------------------------------------------------------------------
// Main scan kernel. 256 threads = 8 independent waves; each wave owns 32
// batch elements and its own LDS staging buffers.
// ---------------------------------------------------------------------------
__global__ void __launch_bounds__(WAVES * 32)
mpsrnn_scan(const int* __restrict__ x,
            const _Float16* __restrict__ Bpack,
            const float* __restrict__ prm,
            float* __restrict__ out)
{
    __shared__ __align__(16) _Float16 aStage[WAVES][32][16];  // h_sel (f16)
    __shared__ __align__(16) float    hBuf[WAVES][32][52];    // comp->batch major

    const int lane  = threadIdx.x & 31;
    const int wave  = threadIdx.x >> 5;
    const int batch = (blockIdx.x * WAVES + wave) * BATCH_PER_WAVE + lane;

    // ---- pack 48 q values (2 bits each) from the spin configuration -------
    unsigned int qp0 = 0u, qp1 = 0u, qp2 = 0u;
    {
        const uint4* xr = reinterpret_cast<const uint4*>(x + (size_t)batch * 96);
        #pragma unroll
        for (int j = 0; j < 24; ++j) {            // 4 ints = 2 sites per load
            const uint4 t = xr[j];
            const unsigned int qa = ((~t.x) >> 31) + 2u * ((~t.y) >> 31);
            const unsigned int qb = ((~t.z) >> 31) + 2u * ((~t.w) >> 31);
            const int sq = 2 * j;
            if      (sq < 16) { qp0 |= qa << (2 * sq);        qp0 |= qb << (2 * sq + 2); }
            else if (sq < 32) { qp1 |= qa << (2 * (sq - 16)); qp1 |= qb << (2 * (sq - 16) + 2); }
            else              { qp2 |= qa << (2 * (sq - 32)); qp2 |= qb << (2 * (sq - 32) + 2); }
        }
    }

    // ---- persistent per-lane state -----------------------------------------
    float hsr[6], hsi[6];
    #pragma unroll
    for (int a = 0; a < 6; ++a) { hsr[a] = 1.0f; hsi[a] = 0.0f; }   // h0 = ones
    float amp = 1.0f;
    float ure = 1.0f, uim = 0.0f;        // accumulated unit phase factor

    _Float16*    aRow = &aStage[wave][lane][0];
    const float* hrow = &hBuf[wave][lane][0];

    for (int s = 0; s < NSITES; ++s) {
        // === publish selected row to LDS as f16 (12 halves) =================
        {
            union { _Float16 h[8]; uint4 u; } lo;
            union { _Float16 h[4]; uint2 u; } hi;
            lo.h[0] = (_Float16)hsr[0]; lo.h[1] = (_Float16)hsr[1];
            lo.h[2] = (_Float16)hsr[2]; lo.h[3] = (_Float16)hsr[3];
            lo.h[4] = (_Float16)hsr[4]; lo.h[5] = (_Float16)hsr[5];
            lo.h[6] = (_Float16)hsi[0]; lo.h[7] = (_Float16)hsi[1];
            hi.h[0] = (_Float16)hsi[2]; hi.h[1] = (_Float16)hsi[3];
            hi.h[2] = (_Float16)hsi[4]; hi.h[3] = (_Float16)hsi[5];
            *reinterpret_cast<uint4*>(aRow)     = lo.u;
            *reinterpret_cast<uint2*>(aRow + 8) = hi.u;
        }

        // === issue all 3 B-tile loads (global) before the LDS fence =========
        const _Float16* bp = Bpack + (size_t)(s * 3) * 512 + lane * 16;
        const v16h Bv0 = *reinterpret_cast<const v16h*>(bp);
        const v16h Bv1 = *reinterpret_cast<const v16h*>(bp + 512);
        const v16h Bv2 = *reinterpret_cast<const v16h*>(bp + 1024);
        if (s + 1 < NSITES)
            __builtin_prefetch(bp + 3 * 512, 0, 3);

        wave_lds_fence();

        // === build A fragments by pure LDS loads (no converts) ==============
        v16h A[2];
        #pragma unroll
        for (int t = 0; t < 2; ++t) {
            const _Float16* src = &aStage[wave][t * 16 + (lane & 15)][0];
            union { v16h v; uint4 q[2]; } av;
            av.q[1] = make_uint4(0u, 0u, 0u, 0u);
            if (lane < 16) {                 // K=0..7: re0..5, im0, im1
                av.q[0] = *reinterpret_cast<const uint4*>(src);
            } else {                         // K=8..15: im2..5, 1.0h (K=12), 0
                const uint2 hi = *reinterpret_cast<const uint2*>(src + 8);
                av.q[0] = make_uint4(hi.x, hi.y, 0x00003C00u, 0u);
            }
            A[t] = av.v;
        }

        // === 6 WMMAs, C = inline 0 ==========================================
        const v8f zc = {};
        v8f D[2][3];
        D[0][0] = __builtin_amdgcn_wmma_f32_16x16x32_f16(false, A[0], false, Bv0, (short)0, zc, false, false);
        D[1][0] = __builtin_amdgcn_wmma_f32_16x16x32_f16(false, A[1], false, Bv0, (short)0, zc, false, false);
        D[0][1] = __builtin_amdgcn_wmma_f32_16x16x32_f16(false, A[0], false, Bv1, (short)0, zc, false, false);
        D[1][1] = __builtin_amdgcn_wmma_f32_16x16x32_f16(false, A[1], false, Bv1, (short)0, zc, false, false);
        D[0][2] = __builtin_amdgcn_wmma_f32_16x16x32_f16(false, A[0], false, Bv2, (short)0, zc, false, false);
        D[1][2] = __builtin_amdgcn_wmma_f32_16x16x32_f16(false, A[1], false, Bv2, (short)0, zc, false, false);

        // === transpose D (comp-per-lane) -> hBuf (batch-per-lane) ===========
        {
            const int nb = lane & 15;
            const int mo = (lane >> 4) * 8;
            #pragma unroll
            for (int t = 0; t < 2; ++t)
                #pragma unroll
                for (int nt = 0; nt < 3; ++nt)
                    #pragma unroll
                    for (int r = 0; r < 8; ++r)
                        hBuf[wave][t * 16 + mo + r][nt * 16 + nb] = D[t][nt][r];
        }

        // === site params (global, independent of LDS): issue pre-fence ======
        const float4* pp = reinterpret_cast<const float4*>(prm + s * 24);
        const float4 p0 = pp[0], p1 = pp[1], p2 = pp[2], p3 = pp[3], p4 = pp[4];

        wave_lds_fence();

        // === per-lane math ===================================================
        const float4* hp = reinterpret_cast<const float4*>(hrow);
        float n2 = 0.0f, acc[4];
        #pragma unroll
        for (int i = 0; i < 4; ++i) {        // row i = 12 contiguous floats
            const float4 u0 = hp[3 * i + 0];
            const float4 u1 = hp[3 * i + 1];
            const float4 u2 = hp[3 * i + 2];
            const float m0 = u0.x * u0.x + u0.y * u0.y;
            const float m1 = u0.z * u0.z + u0.w * u0.w;
            const float m2 = u1.x * u1.x + u1.y * u1.y;
            const float m3 = u1.z * u1.z + u1.w * u1.w;
            const float m4 = u2.x * u2.x + u2.y * u2.y;
            const float m5 = u2.z * u2.z + u2.w * u2.w;
            n2 += (m0 + m1) + (m2 + m3) + (m4 + m5);
            float a0 = m0 * p0.x;
            a0 = fmaf(m1, p0.y, a0); a0 = fmaf(m2, p0.z, a0);
            a0 = fmaf(m3, p0.w, a0); a0 = fmaf(m4, p1.x, a0);
            a0 = fmaf(m5, p1.y, a0);
            acc[i] = a0;                     // eta^2-weighted |h_i|^2 (unnorm.)
        }
        const float inv  = rsqrtf(n2);
        const float sumA = (acc[0] + acc[1]) + (acc[2] + acc[3]);

        const unsigned int qw = (s < 16) ? qp0 : ((s < 32) ? qp1 : qp2);
        const int qc = (int)((qw >> (2 * (s & 15))) & 3u);

        // P[qc]/||P|| == sqrt(acc[qc]/sumA): max- and h-normalizations cancel
        const float a01 = (qc & 1) ? acc[1] : acc[0];
        const float a23 = (qc & 1) ? acc[3] : acc[2];
        const float aq  = (qc & 2) ? a23 : a01;
        amp *= sqrtf(aq) * rsqrtf(sumA);

        // gather row qc via dynamically-addressed LDS read (12 contig floats)
        {
            const float4* hq = reinterpret_cast<const float4*>(hrow + qc * 12);
            const float4 g0 = hq[0], g1 = hq[1], g2 = hq[2];
            hsr[0] = g0.x * inv; hsi[0] = g0.y * inv;
            hsr[1] = g0.z * inv; hsi[1] = g0.w * inv;
            hsr[2] = g1.x * inv; hsi[2] = g1.y * inv;
            hsr[3] = g1.z * inv; hsi[3] = g1.w * inv;
            hsr[4] = g2.x * inv; hsi[4] = g2.y * inv;
            hsr[5] = g2.z * inv; hsi[5] = g2.w * inv;
        }

        // phase: multiply accumulated unit vector by (w.h[qc] + c)/|.|
        {
            // wre = p1.z,p1.w,p2.x..w ; wim = p3.x..w,p4.x,p4.y ; c = p4.z,p4.w
            float dre = p4.z, dim = p4.w;
            dre = fmaf(p1.z, hsr[0], fmaf(-p3.x, hsi[0], dre));
            dim = fmaf(p1.z, hsi[0], fmaf( p3.x, hsr[0], dim));
            dre = fmaf(p1.w, hsr[1], fmaf(-p3.y, hsi[1], dre));
            dim = fmaf(p1.w, hsi[1], fmaf( p3.y, hsr[1], dim));
            dre = fmaf(p2.x, hsr[2], fmaf(-p3.z, hsi[2], dre));
            dim = fmaf(p2.x, hsi[2], fmaf( p3.z, hsr[2], dim));
            dre = fmaf(p2.y, hsr[3], fmaf(-p3.w, hsi[3], dre));
            dim = fmaf(p2.y, hsi[3], fmaf( p3.w, hsr[3], dim));
            dre = fmaf(p2.z, hsr[4], fmaf(-p4.x, hsi[4], dre));
            dim = fmaf(p2.z, hsi[4], fmaf( p4.x, hsr[4], dim));
            dre = fmaf(p2.w, hsr[5], fmaf(-p4.y, hsi[5], dre));
            dim = fmaf(p2.w, hsi[5], fmaf( p4.y, hsr[5], dim));
            const float r2 = dre * dre + dim * dim;
            const float ri = rsqrtf(r2);
            const float zre = (r2 > 0.0f) ? dre * ri : 1.0f;   // angle(0)=0
            const float zim = (r2 > 0.0f) ? dim * ri : 0.0f;
            const float nre = ure * zre - uim * zim;
            uim = fmaf(ure, zim, uim * zre);
            ure = nre;
        }
    }

    // psi = amp * (ure + i*uim), complex64 interleaved
    out[2 * batch + 0] = amp * ure;
    out[2 * batch + 1] = amp * uim;
}

// ---------------------------------------------------------------------------
extern "C" void kernel_launch(void* const* d_in, const int* in_sizes, int n_in,
                              void* d_out, int out_size, void* d_ws, size_t ws_size,
                              hipStream_t stream) {
    (void)in_sizes; (void)n_in; (void)out_size; (void)ws_size;

    const int*   x   = (const int*)  d_in[0];
    const float* M   = (const float*)d_in[1];   // complex64 -> interleaved f32
    const float* v   = (const float*)d_in[2];
    const float* w   = (const float*)d_in[3];
    const float* c   = (const float*)d_in[4];
    const float* eta = (const float*)d_in[5];

    _Float16* Bpack = (_Float16*)((char*)d_ws + WS_BPACK_OFF);
    float*    prm   = (float*)   ((char*)d_ws + WS_PRM_OFF);

    mpsrnn_pack_params<<<NSITES, 256, 0, stream>>>(M, v, w, c, eta, Bpack, prm);

    const int blocks = NBATCH_ / (WAVES * BATCH_PER_WAVE);   // 256
    mpsrnn_scan<<<blocks, WAVES * 32, 0, stream>>>(x, Bpack, prm, (float*)d_out);
}
